// SelfAttention_55009941127398
// MI455X (gfx1250) — matrix operand order repaired and checked by reading, not compiled, not run on hardware
//
#include <hip/hip_runtime.h>
#include <hip/hip_bf16.h>

// ---------------------------------------------------------------------------
// Self-attention B=4, S=4096, D=1024, DK=DV=64 on gfx1250 (MI455X).
//   k0: W fp32[1024][64] -> f16 WT[64][1024] (transposed, so proj B-frags are
//       contiguous 16B runs)
//   k1: fused QKV projection: one A-frag load drives 12 WMMA n-tiles (Q,K,V).
//       Q pre-scaled by 1/sqrt(DK)*log2(e). V stored transposed [B][64][S].
//   k2: flash attention, 128 query rows/block (8 waves x 16 rows).
//       K/V 64-key tiles staged global->LDS with async-to-LDS copies
//       (ASYNCcnt), double buffered so DMA overlaps WMMA. Online softmax in
//       fp32; P transposed C->A layout through per-wave LDS (DScnt).
// ---------------------------------------------------------------------------

typedef _Float16 f16;
typedef __attribute__((ext_vector_type(16))) _Float16 v16h;
typedef __attribute__((ext_vector_type(8)))  _Float16 v8h;
typedef __attribute__((ext_vector_type(8)))  float    v8f;
typedef __attribute__((ext_vector_type(4)))  float    v4f;

#define LOG2E 1.44269504088896340736f
#define LDSPITCH 72   // f16 elements per staged row (64 + 8 pad, 16B aligned)

static __device__ __forceinline__ v16h cat8(v8h lo, v8h hi) {
  return __builtin_shufflevector(lo, hi, 0,1,2,3,4,5,6,7,8,9,10,11,12,13,14,15);
}

// Async global->LDS copy of 16 bytes (per lane). Tracked by ASYNCcnt.
static __device__ __forceinline__ void async_ld16(void* lds_ptr, const void* gptr) {
  unsigned int  loff = (unsigned int)(unsigned long long)(uintptr_t)lds_ptr; // LDS byte addr
  unsigned long long ga = (unsigned long long)(uintptr_t)gptr;
  asm volatile("global_load_async_to_lds_b128 %0, %1, off"
               :: "v"(loff), "v"(ga) : "memory");
}

// ---------------- kernel 0: weight convert + transpose ----------------------
__global__ void wt_kernel(const float* __restrict__ Wq,
                          const float* __restrict__ Wk,
                          const float* __restrict__ Wv,
                          f16* __restrict__ WqT, f16* __restrict__ WkT,
                          f16* __restrict__ WvT) {
  int idx = blockIdx.x * blockDim.x + threadIdx.x;   // 0 .. 3*65536-1
  int m = idx >> 16;
  int i = idx & 65535;
  int d = i >> 6;
  int k = i & 63;
  const float* W = (m == 0) ? Wq : (m == 1) ? Wk : Wv;
  f16* WT        = (m == 0) ? WqT : (m == 1) ? WkT : WvT;
  WT[k * 1024 + d] = (f16)W[d * 64 + k];
}

// ---------------- kernel 1: fused QKV projection (12 WMMA tiles) ------------
// grid: 256 blocks of 64 rows; block: 128 threads (4 waves x 16 rows).
__global__ void __launch_bounds__(128)
proj_kernel(const float* __restrict__ X,
            const f16* __restrict__ WqT, const f16* __restrict__ WkT,
            const f16* __restrict__ WvT,
            f16* __restrict__ Qh, f16* __restrict__ Kh, f16* __restrict__ Vt) {
  const int wave = threadIdx.x >> 5;
  const int lane = threadIdx.x & 31;
  const int half = lane >> 4;
  const int ln   = lane & 15;
  const long r0  = (long)blockIdx.x * 64 + wave * 16;

  v8f acc[12] = {};
  const float* xrow = X + (r0 + ln) * 1024;

  for (int kb = 0; kb < 1024; kb += 32) {
    // A fragment: two contiguous 8-float runs -> f16
    v4f fa0 = *(const v4f*)(xrow + kb + half * 8);
    v4f fa1 = *(const v4f*)(xrow + kb + half * 8 + 4);
    v4f fb0 = *(const v4f*)(xrow + kb + 16 + half * 8);
    v4f fb1 = *(const v4f*)(xrow + kb + 16 + half * 8 + 4);
    v16h a;
#pragma unroll
    for (int j = 0; j < 4; ++j) {
      a[j]      = (f16)fa0[j];
      a[4 + j]  = (f16)fa1[j];
      a[8 + j]  = (f16)fb0[j];
      a[12 + j] = (f16)fb1[j];
    }
    // 12 B fragments: 4 each for Wq, Wk, Wv (compile-time select, no branches)
#pragma unroll
    for (int t = 0; t < 12; ++t) {
      const f16* WT = (t < 4) ? WqT : (t < 8) ? WkT : WvT;
      const f16* wp = WT + (long)((t & 3) * 16 + ln) * 1024 + kb + half * 8;
      v16h bf = cat8(*(const v8h*)wp, *(const v8h*)(wp + 16));
      acc[t] = __builtin_amdgcn_wmma_f32_16x16x32_f16(
          false, a, false, bf, (short)0, acc[t], false, false);
    }
  }

  const float qscale = 0.125f * LOG2E;   // fold softmax scale * log2e into Q
#pragma unroll
  for (int t = 0; t < 4; ++t) {
#pragma unroll
    for (int r = 0; r < 8; ++r) {
      long row = r0 + half * 8 + r;              // C layout: M=half*8+r, N=ln
      int  col = t * 16 + ln;
      long bb  = row >> 12;
      long ss  = row & 4095;
      Qh[row * 64 + col] = (f16)(acc[t][r] * qscale);
      Kh[row * 64 + col] = (f16)acc[t + 4][r];
      Vt[(bb * 64 + col) * 4096 + ss] = (f16)acc[t + 8][r];
    }
  }
}

// ---------------- kernel 2: flash attention (WMMA + async-to-LDS) -----------
// grid: 128 blocks (128 q rows each); block: 256 threads (8 waves x 16 rows)
__global__ void __launch_bounds__(256)
attn_kernel(const f16* __restrict__ Qh, const f16* __restrict__ Kh,
            const f16* __restrict__ Vt, float* __restrict__ Out) {
  __shared__ __align__(16) f16 Kbuf[2][64 * LDSPITCH];   // 2 x 9 KB
  __shared__ __align__(16) f16 Vbuf[2][64 * LDSPITCH];   // 2 x 9 KB
  __shared__ __align__(16) f16 Plds[8][16 * LDSPITCH];   // 18 KB

  const int tid  = threadIdx.x;
  const int wave = tid >> 5;
  const int lane = tid & 31;
  const int half = lane >> 4;
  const int ln   = lane & 15;
  const long r0  = (long)blockIdx.x * 128 + wave * 16;
  const long b   = r0 >> 12;
  const f16* Kb  = Kh + (b << 12) * 64;            // batch base of K
  const f16* Vtb = Vt + b * 64 * 4096;             // batch base of Vt

  // cooperative-copy coordinates: 4 threads per 128B row
  const int crow = tid >> 2;
  const int cseg = (tid & 3) * 16;

  // Q A-fragments (pre-scaled), kept in registers
  const f16* qrow = Qh + (r0 + ln) * 64;
  v16h qa[2];
#pragma unroll
  for (int ks = 0; ks < 2; ++ks) {
    qa[ks] = cat8(*(const v8h*)(qrow + ks * 32 + half * 8),
                  *(const v8h*)(qrow + ks * 32 + 16 + half * 8));
  }

  float mrun[8], lrun[8];
  v8f acc[4] = {};
#pragma unroll
  for (int r = 0; r < 8; ++r) { mrun[r] = -__builtin_inff(); lrun[r] = 0.f; }

  f16* pw = &Plds[wave][0];

  // ---- prefetch tile 0 ----
  {
    const f16* kg = Kb + (long)crow * 64 + cseg;
    const f16* vg = Vtb + (long)crow * 4096 + 0 + cseg;
    f16* kl = &Kbuf[0][crow * LDSPITCH + cseg];
    f16* vl = &Vbuf[0][crow * LDSPITCH + cseg];
    async_ld16(kl, kg);      async_ld16(kl + 8, kg + 8);
    async_ld16(vl, vg);      async_ld16(vl + 8, vg + 8);
  }

  for (int it = 0; it < 64; ++it) {
    const int cur = it & 1;
    asm volatile("s_wait_asynccnt 0" ::: "memory");  // my tile-it copies done
    __syncthreads();                                 // tile it visible block-wide

    if (it + 1 < 64) {                               // prefetch tile it+1
      const int s1 = (it + 1) * 64;
      const f16* kg = Kb + (long)(s1 + crow) * 64 + cseg;
      const f16* vg = Vtb + (long)crow * 4096 + s1 + cseg;
      f16* kl = &Kbuf[cur ^ 1][crow * LDSPITCH + cseg];
      f16* vl = &Vbuf[cur ^ 1][crow * LDSPITCH + cseg];
      async_ld16(kl, kg);    async_ld16(kl + 8, kg + 8);
      async_ld16(vl, vg);    async_ld16(vl + 8, vg + 8);
    }

    const f16* kb = &Kbuf[cur][0];
    const f16* vb = &Vbuf[cur][0];

    // ---- scores: 16 q x 64 keys (log2 domain, scale folded into Q) ----
    v8f sc[4] = {};
#pragma unroll
    for (int t = 0; t < 4; ++t) {
      const f16* kr = kb + (t * 16 + ln) * LDSPITCH;
#pragma unroll
      for (int ks = 0; ks < 2; ++ks) {
        v16h bf = cat8(*(const v8h*)(kr + ks * 32 + half * 8),
                       *(const v8h*)(kr + ks * 32 + 16 + half * 8));
        sc[t] = __builtin_amdgcn_wmma_f32_16x16x32_f16(
            false, qa[ks], false, bf, (short)0, sc[t], false, false);
      }
    }

    // ---- online softmax (rows replicated across each 16-lane half) ----
    float corr[8];
#pragma unroll
    for (int r = 0; r < 8; ++r) {
      float v = fmaxf(fmaxf(sc[0][r], sc[1][r]), fmaxf(sc[2][r], sc[3][r]));
      v = fmaxf(v, __shfl_xor(v, 1, 16));
      v = fmaxf(v, __shfl_xor(v, 2, 16));
      v = fmaxf(v, __shfl_xor(v, 4, 16));
      v = fmaxf(v, __shfl_xor(v, 8, 16));
      float mn = fmaxf(mrun[r], v);
      corr[r]  = exp2f(mrun[r] - mn);
      mrun[r]  = mn;
    }
#pragma unroll
    for (int t = 0; t < 4; ++t)
#pragma unroll
      for (int r = 0; r < 8; ++r)
        sc[t][r] = exp2f(sc[t][r] - mrun[r]);
#pragma unroll
    for (int r = 0; r < 8; ++r) {
      float rs = sc[0][r] + sc[1][r] + sc[2][r] + sc[3][r];
      rs += __shfl_xor(rs, 1, 16);
      rs += __shfl_xor(rs, 2, 16);
      rs += __shfl_xor(rs, 4, 16);
      rs += __shfl_xor(rs, 8, 16);
      lrun[r] = lrun[r] * corr[r] + rs;
#pragma unroll
      for (int t = 0; t < 4; ++t) acc[t][r] *= corr[r];
    }

    // ---- transpose P: C layout -> row-major per-wave LDS ----
#pragma unroll
    for (int t = 0; t < 4; ++t)
#pragma unroll
      for (int r = 0; r < 8; ++r)
        pw[(half * 8 + r) * LDSPITCH + t * 16 + ln] = (f16)sc[t][r];

    asm volatile("s_wait_dscnt 0" ::: "memory");     // same-wave store->load

    // ---- O += P * V  (Vbuf rows are dv, contiguous over key) ----
#pragma unroll
    for (int ks = 0; ks < 2; ++ks) {
      v16h pa = cat8(*(const v8h*)(pw + ln * LDSPITCH + ks * 32 + half * 8),
                     *(const v8h*)(pw + ln * LDSPITCH + ks * 32 + 16 + half * 8));
#pragma unroll
      for (int t = 0; t < 4; ++t) {
        const f16* vr = vb + (t * 16 + ln) * LDSPITCH + ks * 32 + half * 8;
        v16h bv = cat8(*(const v8h*)vr, *(const v8h*)(vr + 16));
        acc[t] = __builtin_amdgcn_wmma_f32_16x16x32_f16(
            false, pa, false, bv, (short)0, acc[t], false, false);
      }
    }
    asm volatile("" ::: "memory");
  }

  // ---- normalize + store fp32 output ----
#pragma unroll
  for (int r = 0; r < 8; ++r) {
    float inv = 1.0f / lrun[r];
    long row  = r0 + half * 8 + r;
#pragma unroll
    for (int t = 0; t < 4; ++t)
      Out[row * 64 + t * 16 + ln] = acc[t][r] * inv;
  }
}

// ---------------------------------------------------------------------------
extern "C" void kernel_launch(void* const* d_in, const int* in_sizes, int n_in,
                              void* d_out, int out_size, void* d_ws, size_t ws_size,
                              hipStream_t stream) {
  const float* X  = (const float*)d_in[0];
  const float* Wq = (const float*)d_in[1];
  const float* Wk = (const float*)d_in[2];
  const float* Wv = (const float*)d_in[3];
  float* Out = (float*)d_out;

  char* ws = (char*)d_ws;
  f16* Qh  = (f16*)(ws);                                // 2 MB
  f16* Kh  = (f16*)(ws + (size_t)2 * 1024 * 1024);      // 2 MB
  f16* Vt  = (f16*)(ws + (size_t)4 * 1024 * 1024);      // 2 MB, [B][64][S]
  f16* WqT = (f16*)(ws + (size_t)6 * 1024 * 1024);      // 128 KB each
  f16* WkT = WqT + 64 * 1024;
  f16* WvT = WkT + 64 * 1024;

  wt_kernel<<<768, 256, 0, stream>>>(Wq, Wk, Wv, WqT, WkT, WvT);
  proj_kernel<<<256, 128, 0, stream>>>(X, WqT, WkT, WvT, Qh, Kh, Vt);
  attn_kernel<<<128, 256, 0, stream>>>(Qh, Kh, Vt, Out);
}
